// NodeEdgeBlock_1640677507363
// MI455X (gfx1250) — compile-verified, matrix operand order checked by reading, not run on hardware
//
#include <hip/hip_runtime.h>
#include <cstddef>
#include <cstdint>

// ---------------------------------------------------------------------------
// MI455X (gfx1250) fused NodeEdgeBlock.
// - Heavy edge GEMMs on v_wmma_f32_16x16x32_bf16 (f32 accumulate).
// - Flash-style online channel-softmax fused into the edge pass (the
//   (4,256,256,256) intermediates never touch HBM).
// - Weights pre-packed into the ISA 16-bit B-fragment VGPR layout, L2-hot.
// - E tiles staged into LDS by the Tensor Data Mover (tensor_load_to_lds,
//   TENSORcnt + s_wait_tensorcnt), issued one j-tile ahead so the DMA
//   overlaps the whole WMMA/softmax phase of the previous tile.
// ---------------------------------------------------------------------------

#define BSZ  4
#define NSEQ 256
#define DXD  256
#define DED  64

#if defined(__has_builtin)
#  if __has_builtin(__builtin_amdgcn_tensor_load_to_lds)
#    define HAVE_TDM 1
#  endif
#endif
#ifndef HAVE_TDM
#  define HAVE_TDM 0
#endif

typedef __attribute__((ext_vector_type(16))) __bf16   v16bf;
typedef __attribute__((ext_vector_type(8)))  float    v8f;
typedef __attribute__((ext_vector_type(4)))  int      v4i;
typedef __attribute__((ext_vector_type(4)))  unsigned v4u;
typedef __attribute__((ext_vector_type(8)))  int      v8i;
typedef __attribute__((ext_vector_type(2)))  float    v2f;

union BFrag { int i[8]; v4i q[2]; v16bf v; };
union FFrag { float f[8]; v8f v; };

static __device__ __forceinline__ unsigned f2bfu(float x) {
  unsigned u = __float_as_uint(x);
  return (u + 0x7FFFu + ((u >> 16) & 1u)) >> 16;   // round-to-nearest-even
}
static __device__ __forceinline__ unsigned packbf(float lo, float hi) {
  return (f2bfu(lo) & 0xFFFFu) | (f2bfu(hi) << 16);
}
static __device__ __forceinline__ float bfu2f(unsigned h) {
  return __uint_as_float(h << 16);
}
static __device__ __forceinline__ void wmma_acc(FFrag& c, const BFrag& a, const BFrag& b) {
  // D = A(16x32 bf16) * B(32x16 bf16) + C(16x16 f32)
  c.v = __builtin_amdgcn_wmma_f32_16x16x32_bf16(false, a.v, false, b.v,
                                                (short)0, c.v, false, false);
}

// 16-bit A/B fragment K mapping (ISA 7.12.2): lane = khalf*16 + (row|col)%16,
// dword v holds K = 2v + 8*khalf + (v>=4 ? 8 : 0)  (per 32-wide k-tile).
static __device__ __forceinline__ int kslot_v(int kk)    { return ((kk & 7) >> 1) + ((kk >> 4) << 2); }
static __device__ __forceinline__ int kslot_half(int kk) { return (kk >> 3) & 1; }

#if HAVE_TDM
// Issue a TDM load of a 32-row x 64-float f32 tile (row stride 64 floats)
// from global into LDS. D# layout per CDNA5 ISA.
// This toolchain's builtin arity: (v4u g0, v8i g1, v4i g2, v4i g3, v8i, i32 cpol).
static __device__ __forceinline__ void tdm_load_tile32x64(const float* gsrc,
                                                          unsigned lds_byte) {
  uint64_t ga = (uint64_t)(uintptr_t)gsrc;
  v4u g0;
  g0[0] = 1u;                                            // count=1 (valid), user mode
  g0[1] = lds_byte;                                      // lds_addr (bytes)
  g0[2] = (unsigned)ga;                                  // global_addr[31:0]
  g0[3] = (unsigned)((ga >> 32) & 0x01FFFFFFu)           // global_addr[56:32]
        | (2u << 30);                                    // type=2 ("image")
  v8i g1;
  g1[0] = 0x00020000;            // workgroup_mask=0, data_size=2 (4B)
  g1[1] = (64 << 16);            // tensor_dim0 = 64 (low16 in bits 63:48)
  g1[2] = 0;                     // tensor_dim0 hi / tensor_dim1 lo
  g1[3] = 1 | (64 << 16);        // tensor_dim1 = 65536 ; tile_dim0 = 64
  g1[4] = 32;                    // tile_dim1 = 32, tile_dim2 = 0
  g1[5] = 64;                    // tensor_dim0_stride = 64 (low 32)
  g1[6] = 0;                     // stride hi / tensor_dim1_stride lo
  g1[7] = 0;
  v4i z4 = {0, 0, 0, 0};                                 // groups 2/3 unused (2D)
  v8i z8 = {0, 0, 0, 0, 0, 0, 0, 0};
  __builtin_amdgcn_tensor_load_to_lds(g0, g1, z4, z4, z8, 0);
}
#endif

// ---------------------------------------------------------------------------
// Pack weight W (dout x din, row-major f32) into bf16 B-fragments:
// dst[((kt*nnt + nt)*32 + lane)*8 + v] = {W[n][K+1], W[n][K]} packed bf16.
// ---------------------------------------------------------------------------
__global__ void pack_bfrag_kernel(const float* __restrict__ W, int din, int nnt,
                                  unsigned* __restrict__ dst, int total) {
  for (int idx = blockIdx.x * blockDim.x + threadIdx.x; idx < total;
       idx += gridDim.x * blockDim.x) {
    int v = idx & 7, lane = (idx >> 3) & 31, rest = idx >> 8;
    int nt = rest % nnt, kt = rest / nnt;
    int n = nt * 16 + (lane & 15);
    int khalf = lane >> 4;
    int K = kt * 32 + 2 * v + 8 * khalf + ((v >= 4) ? 8 : 0);
    dst[idx] = packbf(W[(size_t)n * din + K], W[(size_t)n * din + K + 1]);
  }
}

// ---------------------------------------------------------------------------
// Small y->(4,256) FiLM vectors: ye_add, ye_mul, yx_add, yx_mul.
// ---------------------------------------------------------------------------
__global__ void vec4_kernel(const float* __restrict__ y,
                            const float* __restrict__ Wea, const float* __restrict__ Bea,
                            const float* __restrict__ Wem, const float* __restrict__ Bem,
                            const float* __restrict__ Wxa, const float* __restrict__ Bxa,
                            const float* __restrict__ Wxm, const float* __restrict__ Bxm,
                            float* __restrict__ oea, float* __restrict__ oem,
                            float* __restrict__ oxa, float* __restrict__ oxm) {
  int idx = blockIdx.x * blockDim.x + threadIdx.x;      // 0..4095
  int vec = idx >> 10, b = (idx >> 8) & 3, d = idx & 255;
  const float* W; const float* B; float* O;
  if      (vec == 0) { W = Wea; B = Bea; O = oea; }
  else if (vec == 1) { W = Wem; B = Bem; O = oem; }
  else if (vec == 2) { W = Wxa; B = Bxa; O = oxa; }
  else               { W = Wxm; B = Bxm; O = oxm; }
  float s = B[d];
  for (int k = 0; k < 64; ++k) s += y[b * 64 + k] * W[d * 64 + k];
  O[b * 256 + d] = s;
}

// ---------------------------------------------------------------------------
// Q/K/V projection via WMMA. grid = (32 row-groups, 3 weights), 256 thr.
// Q is pre-scaled by 1/sqrt(DF) and all three are multiplied by node_mask.
// ---------------------------------------------------------------------------
__global__ __launch_bounds__(256)
void qkv_kernel(const float* __restrict__ X, const float* __restrict__ mask,
                const unsigned* __restrict__ Fq, const unsigned* __restrict__ Fk,
                const unsigned* __restrict__ Fv,
                const float* __restrict__ bq, const float* __restrict__ bk,
                const float* __restrict__ bv,
                float* __restrict__ Qs, float* __restrict__ Ks, float* __restrict__ Vs) {
  __shared__ __align__(16) unsigned sA[8192];           // 32 frags (mt*16+kt) x 256 dwords
  const int tid = threadIdx.x, wave = tid >> 5, lane = tid & 31;
  const int row0 = blockIdx.x * 32;
  const int wsel = blockIdx.y;
  const unsigned* F = (wsel == 0) ? Fq : (wsel == 1) ? Fk : Fv;
  const float* bias = (wsel == 0) ? bq : (wsel == 1) ? bk : bv;
  float* out       = (wsel == 0) ? Qs : (wsel == 1) ? Ks : Vs;
  const float scale = (wsel == 0) ? 0.17677669529663688f : 1.0f;   // 1/sqrt(32)

  const v2f* X2 = reinterpret_cast<const v2f*>(X + (size_t)row0 * 256);
#pragma unroll
  for (int u = 0; u < 16; ++u) {
    int idx = tid + 256 * u;                 // 0..4095 (32 rows x 128 pairs)
    int row = idx >> 7, pp = idx & 127, k = pp * 2;
    v2f x2 = X2[row * 128 + pp];
    int mt = row >> 4, kt = k >> 5, kk = k & 31;
    sA[(mt * 16 + kt) * 256 + (kslot_half(kk) * 16 + (row & 15)) * 8 + kslot_v(kk)] =
        packbf(x2.x, x2.y);
  }
  __syncthreads();

  const int dchunk = wave * 32;
  FFrag acc[2][2];
#pragma unroll
  for (int mt = 0; mt < 2; ++mt)
#pragma unroll
    for (int nt = 0; nt < 2; ++nt)
#pragma unroll
      for (int v = 0; v < 8; ++v) acc[mt][nt].f[v] = 0.f;

  for (int kt = 0; kt < 16; ++kt) {
    BFrag a0, a1;
    { const v4i* p = reinterpret_cast<const v4i*>(&sA[(0 * 16 + kt) * 256 + lane * 8]);
      a0.q[0] = p[0]; a0.q[1] = p[1]; }
    { const v4i* p = reinterpret_cast<const v4i*>(&sA[(1 * 16 + kt) * 256 + lane * 8]);
      a1.q[0] = p[0]; a1.q[1] = p[1]; }
#pragma unroll
    for (int nt = 0; nt < 2; ++nt) {
      int ntg = (dchunk >> 4) + nt;
      BFrag bb;
      const v4i* pb = reinterpret_cast<const v4i*>(F + ((size_t)(kt * 16 + ntg) * 32 + lane) * 8);
      __builtin_prefetch(pb + 2, 0, 2);
      bb.q[0] = pb[0]; bb.q[1] = pb[1];
      wmma_acc(acc[0][nt], a0, bb);
      wmma_acc(acc[1][nt], a1, bb);
    }
  }
#pragma unroll
  for (int mt = 0; mt < 2; ++mt)
#pragma unroll
    for (int nt = 0; nt < 2; ++nt)
#pragma unroll
      for (int v = 0; v < 8; ++v) {
        int row = row0 + mt * 16 + v + ((lane >> 4) << 3);
        int d = dchunk + nt * 16 + (lane & 15);
        float val = (acc[mt][nt].f[v] + bias[d]) * mask[row] * scale;
        out[(size_t)row * 256 + d] = val;
      }
}

// ---------------------------------------------------------------------------
// Fused edge kernel. One WG per (b,i); 8 waves; 8 j-tiles of 32.
//  TDM stages E tile (32x64 f32) into LDS one tile ahead (TENSORcnt).
//  GEMM1: Ebf16(32x64) @ {Wmul,Wadd}^T -> E1,E2 (f32)
//  Y = Q*K*(E1+1)+E2 ; online softmax over j per channel d ; wV accumulate
//  Ypre = ye_add + (ye_mul+1)*Y -> bf16 LDS
//  GEMM2: Ypre(32x256) @ Weout^T -> newE tile, masked, stored.
// ---------------------------------------------------------------------------
__global__ __launch_bounds__(256)
void edge_kernel(const float* __restrict__ Qs, const float* __restrict__ Ks,
                 const float* __restrict__ Vs, const float* __restrict__ E,
                 const float* __restrict__ mask,
                 const unsigned* __restrict__ Fmul, const unsigned* __restrict__ Fadd,
                 const unsigned* __restrict__ Feout,
                 const float* __restrict__ bmul, const float* __restrict__ badd,
                 const float* __restrict__ yea, const float* __restrict__ yem,
                 const float* __restrict__ beout,
                 float* __restrict__ outE, float* __restrict__ wV) {
  __shared__ __align__(16) unsigned sEraw[2048];    // 32x64 f32 E tile (TDM dest)
  __shared__ __align__(16) unsigned sEfrag[1024];   // 4 bf16 A-frags (mt,kt)
  __shared__ __align__(16) unsigned sK16[4096];     // 32x256 bf16 K tile
  __shared__ __align__(16) unsigned sV16[4096];     // 32x256 bf16 V tile
  __shared__ __align__(16) unsigned sY16[4096];     // 32x256 bf16 Ypre tile
  __shared__ float sMaskJ[32];

  const float FINF = __builtin_inff();
  const int tid = threadIdx.x, wave = tid >> 5, lane = tid & 31;
  const int bi = blockIdx.x;
  const int b = bi >> 8;
  const float maski = mask[bi];
  const int dchunk = wave * 32;
  const int d0 = dchunk + (lane & 15), d1 = d0 + 16;
  const float q0 = Qs[(size_t)bi * 256 + d0], q1 = Qs[(size_t)bi * 256 + d1];
  const float yea0 = yea[b * 256 + d0], yea1 = yea[b * 256 + d1];
  const float yem0 = yem[b * 256 + d0], yem1 = yem[b * 256 + d1];
  const float bm0 = bmul[d0], bm1 = bmul[d1];
  const float ba0 = badd[d0], ba1 = badd[d1];
  float sm_m[2] = {-FINF, -FINF}, sm_s[2] = {0.f, 0.f}, sm_w[2] = {0.f, 0.f};

#if HAVE_TDM
  const unsigned ldsE = (unsigned)(uintptr_t)(&sEraw[0]);  // LDS byte address
  if (wave == 0)
    tdm_load_tile32x64(E + ((size_t)bi * NSEQ + 0) * DED, ldsE);
#endif

  for (int jt = 0; jt < 8; ++jt) {
    const int j0 = jt * 32;
#if HAVE_TDM
    if (wave == 0) __builtin_amdgcn_s_wait_tensorcnt(0);
#endif
    __syncthreads();
    // ---- P1: E f32 -> bf16 A-frags ; stage K/V (bf16) ; mask -------------
    {
#if HAVE_TDM
      const v2f* E2 = reinterpret_cast<const v2f*>(sEraw);
#else
      const v2f* E2 = reinterpret_cast<const v2f*>(E + ((size_t)bi * NSEQ + j0) * DED);
#endif
#pragma unroll
      for (int u = 0; u < 4; ++u) {
        int idx = tid + 256 * u;             // 0..1023 (32 rows x 32 pairs)
        int row = idx >> 5, kp = idx & 31, k = kp * 2;
        v2f e2 = E2[row * 32 + kp];
        int kt = k >> 5, kk = k & 31;
        int fragid = ((row >> 4) << 1) + kt;
        sEfrag[fragid * 256 + (kslot_half(kk) * 16 + (row & 15)) * 8 + kslot_v(kk)] =
            packbf(e2.x, e2.y);
      }
      const v2f* K2 = reinterpret_cast<const v2f*>(Ks + ((size_t)b * NSEQ + j0) * DXD);
      const v2f* V2 = reinterpret_cast<const v2f*>(Vs + ((size_t)b * NSEQ + j0) * DXD);
#pragma unroll
      for (int u = 0; u < 16; ++u) {
        int idx = tid + 256 * u;             // 0..4095 (32 rows x 128 pairs)
        int row = idx >> 7, pp = idx & 127;
        v2f kv = K2[row * 128 + pp];
        sK16[row * 128 + pp] = packbf(kv.x, kv.y);
        v2f vv = V2[row * 128 + pp];
        sV16[row * 128 + pp] = packbf(vv.x, vv.y);
      }
      if (tid < 32) sMaskJ[tid] = mask[b * 256 + j0 + tid];
    }
    __syncthreads();
#if HAVE_TDM
    // Raw E buffer free now; DMA next tile while we compute this one.
    if (jt < 7 && wave == 0)
      tdm_load_tile32x64(E + ((size_t)bi * NSEQ + (j0 + 32)) * DED, ldsE);
#endif

    // ---- P2: GEMM1 + Y + online softmax + Ypre ---------------------------
    {
      const unsigned short* sKu = reinterpret_cast<const unsigned short*>(sK16);
      const unsigned short* sVu = reinterpret_cast<const unsigned short*>(sV16);
      unsigned short* sYu = reinterpret_cast<unsigned short*>(sY16);
      BFrag afr[4];
#pragma unroll
      for (int f = 0; f < 4; ++f) {
        const v4i* p = reinterpret_cast<const v4i*>(&sEfrag[f * 256 + lane * 8]);
        afr[f].q[0] = p[0]; afr[f].q[1] = p[1];
      }
#pragma unroll
      for (int nt = 0; nt < 2; ++nt) {
        const int ntg = (dchunk >> 4) + nt;
        BFrag bm[2], ba[2];
#pragma unroll
        for (int kt = 0; kt < 2; ++kt) {
          const v4i* pm = reinterpret_cast<const v4i*>(Fmul + ((size_t)(kt * 16 + ntg) * 32 + lane) * 8);
          bm[kt].q[0] = pm[0]; bm[kt].q[1] = pm[1];
          const v4i* pa = reinterpret_cast<const v4i*>(Fadd + ((size_t)(kt * 16 + ntg) * 32 + lane) * 8);
          ba[kt].q[0] = pa[0]; ba[kt].q[1] = pa[1];
        }
        const int d = dchunk + nt * 16 + (lane & 15);
        const float qd = nt ? q1 : q0;
        const float ya = nt ? yea1 : yea0;
        const float ym = nt ? yem1 : yem0;
        const float bmd = nt ? bm1 : bm0;
        const float bad = nt ? ba1 : ba0;
#pragma unroll
        for (int mt = 0; mt < 2; ++mt) {
          FFrag aM, aA;
#pragma unroll
          for (int v = 0; v < 8; ++v) { aM.f[v] = 0.f; aA.f[v] = 0.f; }
#pragma unroll
          for (int kt = 0; kt < 2; ++kt) {
            wmma_acc(aM, afr[mt * 2 + kt], bm[kt]);
            wmma_acc(aA, afr[mt * 2 + kt], ba[kt]);
          }
          float yv[8], vv[8];
          float tmax = -FINF;
#pragma unroll
          for (int v = 0; v < 8; ++v) {
            int jl = mt * 16 + v + ((lane >> 4) << 3);
            float kval = bfu2f(sKu[jl * 256 + d]);
            float vval = bfu2f(sVu[jl * 256 + d]);
            float mj = sMaskJ[jl];
            float mm = maski * mj;
            float e1 = (aM.f[v] + bmd) * mm;
            float e2 = (aA.f[v] + bad) * mm;
            float y = qd * kval * (e1 + 1.f) + e2;     // Q holds mask & 1/sqrt(DF)
            sYu[jl * 256 + d] = (unsigned short)f2bfu(ya + (ym + 1.f) * y);
            float ymk = (mj > 0.f) ? y : -FINF;
            yv[v] = ymk; vv[v] = vval;
            tmax = fmaxf(tmax, ymk);
          }
          tmax = fmaxf(tmax, __shfl_xor(tmax, 16));
          float mold = sm_m[nt];
          float mnew = fmaxf(mold, tmax);
          if (mnew > -FINF) {
            float ps = 0.f, pw = 0.f;
#pragma unroll
            for (int v = 0; v < 8; ++v) {
              float e = __expf(yv[v] - mnew);
              ps += e; pw += e * vv[v];
            }
            ps += __shfl_xor(ps, 16);
            pw += __shfl_xor(pw, 16);
            float sc = (mold > -FINF) ? __expf(mold - mnew) : 0.f;
            sm_s[nt] = sm_s[nt] * sc + ps;
            sm_w[nt] = sm_w[nt] * sc + pw;
            sm_m[nt] = mnew;
          }
        }
      }
    }
    __syncthreads();

    // ---- P3: GEMM2 (Ypre @ Weout^T) -> newE tile -------------------------
    {
      const int mt = wave >> 2, nt = wave & 3;       // 2 m-tiles x 4 n-tiles
      FFrag acc;
#pragma unroll
      for (int v = 0; v < 8; ++v) acc.f[v] = 0.f;
      const int m = lane & 15, khalf = lane >> 4;
      for (int kt = 0; kt < 16; ++kt) {
        BFrag af;
#pragma unroll
        for (int v = 0; v < 8; ++v) {
          int kk = 2 * v + 8 * khalf + ((v >= 4) ? 8 : 0);
          int K = kt * 32 + kk;
          af.i[v] = (int)sY16[(mt * 16 + m) * 128 + (K >> 1)];
        }
        BFrag bf;
        const v4i* pb = reinterpret_cast<const v4i*>(Feout + ((size_t)(kt * 4 + nt) * 32 + lane) * 8);
        __builtin_prefetch(pb + 8, 0, 2);            // 128 B ahead in Feout
        bf.q[0] = pb[0]; bf.q[1] = pb[1];
        wmma_acc(acc, af, bf);
      }
#pragma unroll
      for (int v = 0; v < 8; ++v) {
        int jl = mt * 16 + v + ((lane >> 4) << 3);
        int eo = nt * 16 + (lane & 15);
        float val = (acc.f[v] + beout[eo]) * maski * sMaskJ[jl];
        outE[((size_t)bi * NSEQ + (j0 + jl)) * DED + eo] = val;
      }
    }
  }

  // ---- finalize wV = softmax-weighted V --------------------------------
  if (lane < 16) {
    float w0 = (sm_s[0] > 0.f) ? sm_w[0] / sm_s[0] : 0.f;
    float w1 = (sm_s[1] > 0.f) ? sm_w[1] / sm_s[1] : 0.f;
    wV[(size_t)bi * 256 + d0] = w0;
    wV[(size_t)bi * 256 + d1] = w1;
  }
}

// ---------------------------------------------------------------------------
// newX = x_out( yx_add + (yx_mul+1)*wV ) * mask  via WMMA.
// ---------------------------------------------------------------------------
__global__ __launch_bounds__(256)
void xout_kernel(const float* __restrict__ wV, const float* __restrict__ yxa,
                 const float* __restrict__ yxm, const float* __restrict__ mask,
                 const unsigned* __restrict__ Fx, const float* __restrict__ bx,
                 float* __restrict__ outX) {
  __shared__ __align__(16) unsigned sA[8192];
  const int tid = threadIdx.x, wave = tid >> 5, lane = tid & 31;
  const int row0 = blockIdx.x * 32;
#pragma unroll
  for (int u = 0; u < 16; ++u) {
    int idx = tid + 256 * u;
    int row = idx >> 7, pp = idx & 127, k = pp * 2;
    int grow = row0 + row, bb = grow >> 8;
    v2f wv2 = reinterpret_cast<const v2f*>(wV + (size_t)grow * 256)[pp];
    v2f a2  = reinterpret_cast<const v2f*>(yxa + bb * 256)[pp];
    v2f m2  = reinterpret_cast<const v2f*>(yxm + bb * 256)[pp];
    float u0 = a2.x + (m2.x + 1.f) * wv2.x;
    float u1 = a2.y + (m2.y + 1.f) * wv2.y;
    int mt = row >> 4, kt = k >> 5, kk = k & 31;
    sA[(mt * 16 + kt) * 256 + (kslot_half(kk) * 16 + (row & 15)) * 8 + kslot_v(kk)] =
        packbf(u0, u1);
  }
  __syncthreads();

  const int dchunk = wave * 32;
  FFrag acc[2][2];
#pragma unroll
  for (int mt = 0; mt < 2; ++mt)
#pragma unroll
    for (int nt = 0; nt < 2; ++nt)
#pragma unroll
      for (int v = 0; v < 8; ++v) acc[mt][nt].f[v] = 0.f;

  for (int kt = 0; kt < 16; ++kt) {
    BFrag a0, a1;
    { const v4i* p = reinterpret_cast<const v4i*>(&sA[(0 * 16 + kt) * 256 + lane * 8]);
      a0.q[0] = p[0]; a0.q[1] = p[1]; }
    { const v4i* p = reinterpret_cast<const v4i*>(&sA[(1 * 16 + kt) * 256 + lane * 8]);
      a1.q[0] = p[0]; a1.q[1] = p[1]; }
#pragma unroll
    for (int nt = 0; nt < 2; ++nt) {
      int ntg = (dchunk >> 4) + nt;
      BFrag bb;
      const v4i* pb = reinterpret_cast<const v4i*>(Fx + ((size_t)(kt * 16 + ntg) * 32 + lane) * 8);
      bb.q[0] = pb[0]; bb.q[1] = pb[1];
      wmma_acc(acc[0][nt], a0, bb);
      wmma_acc(acc[1][nt], a1, bb);
    }
  }
#pragma unroll
  for (int mt = 0; mt < 2; ++mt)
#pragma unroll
    for (int nt = 0; nt < 2; ++nt)
#pragma unroll
      for (int v = 0; v < 8; ++v) {
        int row = row0 + mt * 16 + v + ((lane >> 4) << 3);
        int d = dchunk + nt * 16 + (lane & 15);
        outX[(size_t)row * 256 + d] = (acc[mt][nt].f[v] + bx[d]) * mask[row];
      }
}

// ---------------------------------------------------------------------------
// Stats + final y path (tiny).
// ---------------------------------------------------------------------------
__global__ void xstats_kernel(const float* __restrict__ X, float* __restrict__ xz) {
  const float FINF = __builtin_inff();
  int idx = blockIdx.x * blockDim.x + threadIdx.x;  // 0..1023
  int b = idx >> 8, d = idx & 255;
  float s = 0.f, q = 0.f, mn = FINF, mx = -FINF;
  for (int j = 0; j < 256; ++j) {
    float v = X[((size_t)b * 256 + j) * 256 + d];
    s += v; q += v * v; mn = fminf(mn, v); mx = fmaxf(mx, v);
  }
  float mean = s / 256.f;
  float var = (q - 256.f * mean * mean) / 255.f;
  xz[b * 1024 + d]       = mean;
  xz[b * 1024 + 256 + d] = mn;
  xz[b * 1024 + 512 + d] = mx;
  xz[b * 1024 + 768 + d] = sqrtf(fmaxf(var, 0.f));
}

__global__ __launch_bounds__(256)
void estats_kernel(const float* __restrict__ E, float* __restrict__ ez) {
  const float FINF = __builtin_inff();
  int bc = blockIdx.x;                 // 0..255
  int b = bc >> 6, c = bc & 63;
  int tid = threadIdx.x;
  float s = 0.f, q = 0.f, mn = FINF, mx = -FINF;
  for (int t = tid; t < 65536; t += 256) {
    float v = E[((size_t)b * 65536 + t) * 64 + c];
    s += v; q += v * v; mn = fminf(mn, v); mx = fmaxf(mx, v);
  }
  __shared__ float rs[256], rq[256], rn[256], rx[256];
  rs[tid] = s; rq[tid] = q; rn[tid] = mn; rx[tid] = mx;
  __syncthreads();
  for (int off = 128; off > 0; off >>= 1) {
    if (tid < off) {
      rs[tid] += rs[tid + off]; rq[tid] += rq[tid + off];
      rn[tid] = fminf(rn[tid], rn[tid + off]);
      rx[tid] = fmaxf(rx[tid], rx[tid + off]);
    }
    __syncthreads();
  }
  if (tid == 0) {
    float n = 65536.f;
    float mean = rs[0] / n;
    float var = (rq[0] - n * mean * mean) / (n - 1.f);
    ez[b * 256 + c]        = mean;
    ez[b * 256 + 64 + c]   = rn[0];
    ez[b * 256 + 128 + c]  = rx[0];
    ez[b * 256 + 192 + c]  = sqrtf(fmaxf(var, 0.f));
  }
}

__global__ __launch_bounds__(64)
void finaly_kernel(const float* __restrict__ y, const float* __restrict__ xz,
                   const float* __restrict__ ez,
                   const float* __restrict__ Wyy, const float* __restrict__ Byy,
                   const float* __restrict__ Wxy, const float* __restrict__ Bxy,
                   const float* __restrict__ Wey, const float* __restrict__ Bey,
                   const float* __restrict__ W1, const float* __restrict__ B1,
                   const float* __restrict__ W2, const float* __restrict__ B2,
                   float* __restrict__ outY) {
  __shared__ float sN[64], sH[64];
  int b = blockIdx.x, d = threadIdx.x;
  float s1 = Byy[d];
  for (int k = 0; k < 64; ++k) s1 += y[b * 64 + k] * Wyy[d * 64 + k];
  float s2 = Bxy[d];
  for (int k = 0; k < 1024; ++k) s2 += xz[b * 1024 + k] * Wxy[d * 1024 + k];
  float s3 = Bey[d];
  for (int k = 0; k < 256; ++k) s3 += ez[b * 256 + k] * Wey[d * 256 + k];
  sN[d] = s1 + s2 + s3;
  __syncthreads();
  float h = B1[d];
  for (int k = 0; k < 64; ++k) h += sN[k] * W1[d * 64 + k];
  sH[d] = fmaxf(h, 0.f);
  __syncthreads();
  float o = B2[d];
  for (int k = 0; k < 64; ++k) o += sH[k] * W2[d * 64 + k];
  outY[b * 64 + d] = o;
}

// ---------------------------------------------------------------------------
extern "C" void kernel_launch(void* const* d_in, const int* in_sizes, int n_in,
                              void* d_out, int out_size, void* d_ws, size_t ws_size,
                              hipStream_t stream) {
  (void)in_sizes; (void)n_in; (void)out_size; (void)ws_size;
  const float* X     = (const float*)d_in[0];
  const float* E     = (const float*)d_in[1];
  const float* y     = (const float*)d_in[2];
  const float* mask  = (const float*)d_in[3];
  const float* qw    = (const float*)d_in[4];  const float* qb   = (const float*)d_in[5];
  const float* kw    = (const float*)d_in[6];  const float* kb   = (const float*)d_in[7];
  const float* vw    = (const float*)d_in[8];  const float* vb   = (const float*)d_in[9];
  const float* eaddw = (const float*)d_in[10]; const float* eaddb = (const float*)d_in[11];
  const float* emulw = (const float*)d_in[12]; const float* emulb = (const float*)d_in[13];
  const float* yemw  = (const float*)d_in[14]; const float* yemb = (const float*)d_in[15];
  const float* yeaw  = (const float*)d_in[16]; const float* yeab = (const float*)d_in[17];
  const float* yxmw  = (const float*)d_in[18]; const float* yxmb = (const float*)d_in[19];
  const float* yxaw  = (const float*)d_in[20]; const float* yxab = (const float*)d_in[21];
  const float* yyw   = (const float*)d_in[22]; const float* yyb  = (const float*)d_in[23];
  const float* xyw   = (const float*)d_in[24]; const float* xyb  = (const float*)d_in[25];
  const float* eyw   = (const float*)d_in[26]; const float* eyb  = (const float*)d_in[27];
  const float* xow   = (const float*)d_in[28]; const float* xob  = (const float*)d_in[29];
  const float* eow   = (const float*)d_in[30]; const float* eob  = (const float*)d_in[31];
  const float* y1w   = (const float*)d_in[32]; const float* y1b  = (const float*)d_in[33];
  const float* y2w   = (const float*)d_in[34]; const float* y2b  = (const float*)d_in[35];

  float* outX = (float*)d_out;                         // (4,256,256)
  float* outE = outX + (size_t)BSZ * NSEQ * NSEQ;      // (4,256,256,64)
  float* outY = outX + 17039360;                       // (4,64)

  char* wsb = (char*)d_ws;
  float* Qs   = (float*)(wsb + (0u << 20));
  float* Ks   = (float*)(wsb + (1u << 20));
  float* Vs   = (float*)(wsb + (2u << 20));
  float* wVb  = (float*)(wsb + (3u << 20));
  float* vyea = (float*)(wsb + (4u << 20));
  float* vyem = vyea + 1024;
  float* vyxa = vyem + 1024;
  float* vyxm = vyxa + 1024;
  float* xz   = vyxm + 1024;     // 4096 floats
  float* ez   = xz + 4096;       // 1024 floats
  unsigned* Fq    = (unsigned*)(ez + 1024);
  unsigned* Fk    = Fq + 65536;
  unsigned* Fv    = Fk + 65536;
  unsigned* Fx    = Fv + 65536;
  unsigned* Fmul  = Fx + 65536;
  unsigned* Fadd  = Fmul + 8192;
  unsigned* Feout = Fadd + 8192;

  // 1) pack weights into bf16 B-fragments (L2-resident for all GEMMs)
  pack_bfrag_kernel<<<256, 256, 0, stream>>>(qw,    256, 16, Fq,    65536);
  pack_bfrag_kernel<<<256, 256, 0, stream>>>(kw,    256, 16, Fk,    65536);
  pack_bfrag_kernel<<<256, 256, 0, stream>>>(vw,    256, 16, Fv,    65536);
  pack_bfrag_kernel<<<256, 256, 0, stream>>>(xow,   256, 16, Fx,    65536);
  pack_bfrag_kernel<<<32,  256, 0, stream>>>(emulw,  64, 16, Fmul,  8192);
  pack_bfrag_kernel<<<32,  256, 0, stream>>>(eaddw,  64, 16, Fadd,  8192);
  pack_bfrag_kernel<<<64,  256, 0, stream>>>(eow,   256,  4, Feout, 16384);

  // 2) FiLM vectors from y
  vec4_kernel<<<16, 256, 0, stream>>>(y, yeaw, yeab, yemw, yemb,
                                      yxaw, yxab, yxmw, yxmb,
                                      vyea, vyem, vyxa, vyxm);

  // 3) Q/K/V projections (WMMA)
  qkv_kernel<<<dim3(32, 3), 256, 0, stream>>>(X, mask, Fq, Fk, Fv,
                                              qb, kb, vb, Qs, Ks, Vs);

  // 4) fused edge pass: newE + online-softmax wV (WMMA + TDM staging)
  edge_kernel<<<BSZ * NSEQ, 256, 0, stream>>>(Qs, Ks, Vs, E, mask,
                                              Fmul, Fadd, Feout,
                                              emulb, eaddb, vyea, vyem, eob,
                                              outE, wVb);

  // 5) newX (WMMA)
  xout_kernel<<<32, 256, 0, stream>>>(wVb, vyxa, vyxm, mask, Fx, xob, outX);

  // 6) y path
  xstats_kernel<<<4, 256, 0, stream>>>(X, xz);
  estats_kernel<<<256, 256, 0, stream>>>(E, ez);
  finaly_kernel<<<4, 64, 0, stream>>>(y, xz, ez, yyw, yyb, xyw, xyb,
                                      eyw, eyb, y1w, y1b, y2w, y2b, outY);
}
